// STU_7301444403345
// MI455X (gfx1250) — matrix operand order repaired and checked by reading, not compiled, not run on hardware
//
#include <hip/hip_runtime.h>
#include <hip/hip_bf16.h>

typedef __bf16 bf16;
typedef __attribute__((ext_vector_type(16))) __bf16 v16bf;
typedef __attribute__((ext_vector_type(8)))  __bf16 v8bf;
typedef __attribute__((ext_vector_type(8)))  float  v8f;
typedef __attribute__((ext_vector_type(4)))  unsigned int u32x4;
typedef __attribute__((ext_vector_type(8)))  unsigned int u32x8;

#define D_OUT 256
#define SEQ   2048
#define KEIG  24
#define KU    3
#define KY    2
#define BATCH 8
#define KDIM  (KEIG * D_OUT)   // 6144
#define YSTR  520              // padded LDS row stride for recurrence
#define USTR  40               // padded LDS row stride for u^T slab (64B row + 16B pad, matches TDM pad)
#define USE_TDM 1

static __device__ __forceinline__ bf16 tobf(float f) { return (bf16)f; }

static __device__ __forceinline__ v16bf combine16(v8bf lo, v8bf hi) {
    v16bf r;
#pragma unroll
    for (int i = 0; i < 8; ++i) { r[i] = lo[i]; r[i + 8] = hi[i]; }
    return r;
}
static __device__ __forceinline__ v16bf ld16(const bf16* p) {          // 32B contiguous, 16B aligned
    return combine16(*(const v8bf*)p, *(const v8bf*)(p + 8));
}
static __device__ __forceinline__ v16bf ld16_split(const bf16* p0, const bf16* p1) {
    return combine16(*(const v8bf*)p0, *(const v8bf*)p1);
}
static __device__ __forceinline__ v8f wmma_bf16(v16bf a, v16bf b, v8f c) {
    return __builtin_amdgcn_wmma_f32_16x16x32_bf16(false, a, false, b, (short)0, c, false, false);
}

static __device__ __forceinline__ void wait_tensor0() {
#if __has_builtin(__builtin_amdgcn_s_wait_tensorcnt)
    __builtin_amdgcn_s_wait_tensorcnt(0);
#else
    asm volatile("s_wait_tensorcnt 0x0" ::: "memory");
#endif
}

#if USE_TDM
// TDM: DMA a 2D tile [tile_rows x 32 bf16] (row stride SEQ elements) from global into LDS,
// with hardware padding: 64B (one row) stored -> 16B pad, giving USTR(=40)-element LDS rows.
static __device__ __forceinline__ void tdm_issue_slab(unsigned lds_off, const bf16* gsrc) {
    unsigned long long ga = (unsigned long long)(uintptr_t)gsrc;
    u32x4 g0;
    g0[0] = 1u;                                                   // count=1, user descriptor
    g0[1] = lds_off;                                              // lds_addr
    g0[2] = (unsigned)ga;                                         // global_addr[31:0]
    g0[3] = ((unsigned)(ga >> 32) & 0x01FFFFFFu) | (2u << 30);    // global_addr[56:32] | type=2
    u32x8 g1;
    g1[0] = (1u << 16)        // data_size = 2 bytes
          | (1u << 20)        // pad_enable
          | (3u << 22)        // pad_interval: 16 DWORDs (64B) between pads
          | (3u << 25);       // pad_amount: 4 DWORDs (16B)
    g1[1] = ((unsigned)SEQ & 0xFFFFu) << 16;     // tensor_dim0[15:0]
    g1[2] = ((unsigned)D_OUT & 0xFFFFu) << 16;   // tensor_dim1[15:0]
    g1[3] = (32u & 0xFFFFu) << 16;               // tile_dim0 = 32
    g1[4] = (unsigned)D_OUT;                     // tile_dim1 = 256, tile_dim2 = 0
    g1[5] = (unsigned)SEQ;                       // tensor_dim0_stride[31:0]
    g1[6] = 0u;
    g1[7] = 0u;
    u32x4 gz;
    gz[0] = 0u; gz[1] = 0u; gz[2] = 0u; gz[3] = 0u;
    asm volatile("tensor_load_to_lds %0, %1, %2, %3"
                 :: "s"(g0), "s"(g1), "s"(gz), "s"(gz)
                 : "memory");
}
#endif

// ---------------------------------------------------------------- prep kernels
__global__ void prep_filters(const float* __restrict__ eig_vecs,
                             const float* __restrict__ eig_vals,
                             bf16* __restrict__ vS) {          // [2048][24], pre-scaled by lambda^0.25
    int idx = blockIdx.x * blockDim.x + threadIdx.x;
    if (idx >= SEQ * KEIG) return;
    int f = idx % KEIG;
    vS[idx] = tobf(eig_vecs[idx] * powf(eig_vals[f], 0.25f));
}

__global__ void prep_phiT(const float* __restrict__ m_phi,    // [6144][256]
                          bf16* __restrict__ phiT) {          // [256][6144]
    int idx = blockIdx.x * blockDim.x + threadIdx.x;
    if (idx >= D_OUT * KDIM) return;
    int o = idx / KDIM, k = idx % KDIM;
    phiT[idx] = tobf(m_phi[k * D_OUT + o]);
}

__global__ void prep_mu(const float* __restrict__ m_u,        // [256][256][3]
                        bf16* __restrict__ muB) {             // [3][256 o][256 c]
    int idx = blockIdx.x * blockDim.x + threadIdx.x;
    if (idx >= KU * D_OUT * D_OUT) return;
    int i = idx / (D_OUT * D_OUT);
    int o = (idx / D_OUT) % D_OUT;
    int c = idx % D_OUT;
    muB[idx] = tobf(m_u[(o * D_OUT + c) * KU + i]);
}

__global__ void prep_my(const float* __restrict__ m_y,        // [256][2][256]
                        bf16* __restrict__ myB) {             // [2][256 o][256 c]
    int idx = blockIdx.x * blockDim.x + threadIdx.x;
    if (idx >= KY * D_OUT * D_OUT) return;
    int j = idx / (D_OUT * D_OUT);
    int o = (idx / D_OUT) % D_OUT;
    int c = idx % D_OUT;
    myB[idx] = tobf(m_y[o * (KY * D_OUT) + j * D_OUT + c]);
}

// tiled transpose: u[b][s][c] f32 -> uT[b][c][s] bf16 + uP[b][s][c] bf16
__global__ __launch_bounds__(256) void prep_u(const float* __restrict__ u,
                                              bf16* __restrict__ uT,
                                              bf16* __restrict__ uP) {
    __shared__ float tile[32][33];
    int b  = blockIdx.z;
    int s0 = blockIdx.x * 32;
    int c0 = blockIdx.y * 32;
    int tx = threadIdx.x & 31;
    int ty = threadIdx.x >> 5;
    for (int r = ty; r < 32; r += 8) {
        float v = u[((size_t)b * SEQ + s0 + r) * D_OUT + c0 + tx];
        tile[r][tx] = v;
        uP[((size_t)b * SEQ + s0 + r) * D_OUT + c0 + tx] = tobf(v);
    }
    __syncthreads();
    for (int r = ty; r < 32; r += 8)
        uT[((size_t)b * D_OUT + c0 + r) * SEQ + s0 + tx] = tobf(tile[tx][r]);
}

// ------------------------------------------------- stage 1: triangular Toeplitz conv
// out[t,f,c] = sum_{s'<=t} v[t-s', f] * u[s', c], per batch.
// Grid: (tt=128, fg=3). Block 256 (8 waves): wave w -> 2 filters (fg*8 + (w&3)*2 + {0,1}),
// column-half w>>2 (8 N-tiles). 16 accumulators per wave; B tiles feed 2 WMMAs each.
// u^T slab staged by the Tensor Data Mover, double-buffered to overlap DMA with WMMA.
__global__ __launch_bounds__(256) void conv_kernel(const bf16* __restrict__ uTb,  // [256][2048] (this batch)
                                                   const bf16* __restrict__ vS,   // [2048][24] scaled
                                                   bf16* __restrict__ xt) {       // [2048][6144] (this batch)
    __shared__ bf16 uTl[2][D_OUT * USTR];   // 2 x 20KB, TDM-padded rows
    __shared__ bf16 vaL[2][8 * 48];         // reversed filter windows, 8 filters, double-buffered

    const int tt    = blockIdx.x;
    const int fg    = blockIdx.y;
    const int tid   = threadIdx.x;
    const int lane  = tid & 31, w = tid >> 5;
    const int m     = lane & 15, hi = lane >> 4;
    const int fpair = w & 3, chalf = w >> 2;
    const int t0    = tt * 16;
    const int nk    = tt / 2 + 1;           // K-chunks covering s' in [0, t0+15]

    v8f zero = {};
    v8f acc[2][8];
#pragma unroll
    for (int fi = 0; fi < 2; ++fi)
#pragma unroll
        for (int nt = 0; nt < 8; ++nt) acc[fi][nt] = zero;

    // ---- stage filter windows for a chunk: va[fs][j2] = v[t0 - s0 + 15 - j2] (masks s' > t)
    auto stage_va = [&](int buf, int s0) {
        for (int i = tid; i < 8 * 48; i += 256) {
            int fs2 = i / 48, j2 = i % 48;
            int g = t0 - s0 + 15 - j2;
            vaL[buf][i] = (g >= 0) ? vS[g * KEIG + fg * 8 + fs2] : tobf(0.0f);
        }
    };
#if !USE_TDM
    auto stage_u = [&](int buf, int s0) {   // fallback: manual staging
        const int c = tid;
        const uint4* g = (const uint4*)(uTb + c * SEQ + s0);
        uint4* dst = (uint4*)(&uTl[buf][c * USTR]);
        dst[0] = g[0]; dst[1] = g[1]; dst[2] = g[2]; dst[3] = g[3];
    };
#endif

    // ---- prologue: fill buffer 0
#if USE_TDM
    if (w == 0) {
        tdm_issue_slab((unsigned)(uintptr_t)&uTl[0][0], uTb);
    }
    stage_va(0, 0);
    if (w == 0) wait_tensor0();
#else
    stage_u(0, 0);
    stage_va(0, 0);
#endif
    __syncthreads();

    for (int kc = 0; kc < nk; ++kc) {
        const int cur = kc & 1;
        // kick off next chunk's staging (overlaps with this chunk's WMMAs)
        if (kc + 1 < nk) {
#if USE_TDM
            if (w == 0) {
                tdm_issue_slab((unsigned)(uintptr_t)&uTl[cur ^ 1][0], uTb + (kc + 1) * 32);
            }
#else
            stage_u(cur ^ 1, (kc + 1) * 32);
#endif
            stage_va(cur ^ 1, (kc + 1) * 32);
        }

        // A fragments (16x32 bf16): A[m][K] = va[15 - m + K]; 2 filters
        v16bf a[2];
#pragma unroll
        for (int fi = 0; fi < 2; ++fi) {
#pragma unroll
            for (int e = 0; e < 16; ++e) {
                int K = hi * 16 + e;
                a[fi][e] = vaL[cur][(fpair * 2 + fi) * 48 + 15 - m + K];
            }
        }
#pragma unroll
        for (int nt = 0; nt < 8; ++nt) {    // B loaded once, feeds 2 WMMAs
            const bf16* bp = &uTl[cur][(chalf * 128 + nt * 16 + m) * USTR + hi * 16];
            v16bf b = ld16(bp);
            acc[0][nt] = wmma_bf16(a[0], b, acc[0][nt]);
            acc[1][nt] = wmma_bf16(a[1], b, acc[1][nt]);
        }

#if USE_TDM
        if (w == 0 && kc + 1 < nk) wait_tensor0();
#endif
        __syncthreads();
    }

    // D layout: lane l, vgpr r -> M = r + hi*8, N = l&15
#pragma unroll
    for (int fi = 0; fi < 2; ++fi) {
        int f = fg * 8 + fpair * 2 + fi;
#pragma unroll
        for (int nt = 0; nt < 8; ++nt) {
            int c = chalf * 128 + nt * 16 + m;
#pragma unroll
            for (int r = 0; r < 8; ++r) {
                int t = t0 + r + hi * 8;
                xt[(size_t)t * KDIM + f * D_OUT + c] = tobf(acc[fi][nt][r]);
            }
        }
    }
}

// --------------------------- stage 2: delta = x_tilde @ m_phi + sum_i shift_i(u) @ m_u[:,:,i]^T
// Grid: 64 t-tiles of 32 rows. Block 256 (8 waves): wave -> 2 M-tiles x 2 N-tiles;
// each B fragment feeds 2 WMMAs. K = 6144 + 3*256.
__global__ __launch_bounds__(256) void proj_kernel(const bf16* __restrict__ xt,    // [2048][6144]
                                                   const bf16* __restrict__ uPb,   // [2048][256]
                                                   const bf16* __restrict__ phiT,  // [256][6144]
                                                   const bf16* __restrict__ muB,   // [3][256][256]
                                                   float* __restrict__ delta) {    // [2048][256]
    __shared__ bf16 xta[32 * 32];
    const int tid  = threadIdx.x;
    const int lane = tid & 31, w = tid >> 5;
    const int m    = lane & 15, hi = lane >> 4;
    const int t0   = blockIdx.x * 32;
    const int o0   = (w * 2) * 16 + m;
    const int o1   = (w * 2 + 1) * 16 + m;

    v8f acc[2][2];
    v8f zero = {};
    acc[0][0] = zero; acc[0][1] = zero; acc[1][0] = zero; acc[1][1] = zero;

    // spectral K loop
    for (int kc = 0; kc < KDIM / 32; ++kc) {
        int k0 = kc * 32;
        __syncthreads();
#pragma unroll
        for (int p = 0; p < 2; ++p) {       // stage 32x32 A tile (one u32 = 2 bf16 per slot)
            int lin = tid + p * 256;
            int row = lin >> 4, colp = (lin & 15) * 2;
            *(unsigned*)(xta + row * 32 + colp) =
                *(const unsigned*)(xt + (size_t)(t0 + row) * KDIM + k0 + colp);
        }
        __syncthreads();
        v16bf b0 = ld16(phiT + (size_t)o0 * KDIM + k0 + hi * 16);
        v16bf b1 = ld16(phiT + (size_t)o1 * KDIM + k0 + hi * 16);
#pragma unroll
        for (int mt = 0; mt < 2; ++mt) {
            const bf16* ab = xta + (mt * 16 + m) * 32 + hi * 16;
            v16bf a = ld16(ab);
            acc[mt][0] = wmma_bf16(a, b0, acc[mt][0]);
            acc[mt][1] = wmma_bf16(a, b1, acc[mt][1]);
        }
    }
    // autoregressive K loop (3 shifted copies of u)
    for (int i = 0; i < KU; ++i) {
        for (int kc = 0; kc < D_OUT / 32; ++kc) {
            int k0 = kc * 32;
            __syncthreads();
#pragma unroll
            for (int p = 0; p < 2; ++p) {
                int lin = tid + p * 256;
                int row = lin >> 4, colp = (lin & 15) * 2;
                int trow = t0 + row - i;
                unsigned val = 0u;
                if (trow >= 0) val = *(const unsigned*)(uPb + (size_t)trow * D_OUT + k0 + colp);
                *(unsigned*)(xta + row * 32 + colp) = val;
            }
            __syncthreads();
            v16bf b0 = ld16(muB + ((size_t)i * D_OUT + o0) * D_OUT + k0 + hi * 16);
            v16bf b1 = ld16(muB + ((size_t)i * D_OUT + o1) * D_OUT + k0 + hi * 16);
#pragma unroll
            for (int mt = 0; mt < 2; ++mt) {
                const bf16* ab = xta + (mt * 16 + m) * 32 + hi * 16;
                v16bf a = ld16(ab);
                acc[mt][0] = wmma_bf16(a, b0, acc[mt][0]);
                acc[mt][1] = wmma_bf16(a, b1, acc[mt][1]);
            }
        }
    }
#pragma unroll
    for (int mt = 0; mt < 2; ++mt)
#pragma unroll
        for (int r = 0; r < 8; ++r) {
            int t = t0 + mt * 16 + r + hi * 8;
            delta[(size_t)t * D_OUT + o0] = acc[mt][0][r];
            delta[(size_t)t * D_OUT + o1] = acc[mt][1][r];
        }
}

// --------------------------- stage 3: y_t = M0 y_{t-1} + M1 y_{t-2} + delta_t (batch-stacked WMMA)
// One WG, 512 threads = 16 waves = 16 output N-tiles. M dim = batch (rows 0..7), K = 512 (Y1||Y2).
// delta is software-pipelined: delta[t+1] loads issue after y_t is written, hiding global latency
// under the next step's 16 WMMAs.
__global__ __launch_bounds__(512) void recur_kernel(const bf16* __restrict__ myB,   // [2][256][256]
                                                    const float* __restrict__ delta,// [8][2048][256]
                                                    float* __restrict__ out) {      // [8][2048][256]
    __shared__ bf16 Yb[2 * 16 * YSTR];
    const int tid  = threadIdx.x;
    const int lane = tid & 31, nt = tid >> 5;
    const int m    = lane & 15, hi = lane >> 4;
    const int o    = nt * 16 + m;

    for (int i = tid; i < 2 * 16 * YSTR; i += 512) Yb[i] = tobf(0.0f);

    // preload all 16 B fragments of [M0; M1]
    v16bf Breg[16];
#pragma unroll
    for (int kc = 0; kc < 16; ++kc) {
        int j  = kc >> 3;
        int c0 = (kc & 7) * 32;
        Breg[kc] = ld16(myB + ((size_t)j * D_OUT + o) * D_OUT + c0 + hi * 16);
    }

    float dreg[8];
    if (hi == 0) {
#pragma unroll
        for (int r = 0; r < 8; ++r) dreg[r] = delta[((size_t)r * SEQ + 0) * D_OUT + o];
    }
    __syncthreads();

    for (int t = 0; t < SEQ; ++t) {
        v8f acc = {};
#pragma unroll
        for (int kc = 0; kc < 16; ++kc) {
            // kc<8 reads y_{t-1} (slot (t+1)&1), kc>=8 reads y_{t-2} (slot t&1)
            int slot = (kc < 8) ? ((t + 1) & 1) : (t & 1);
            const bf16* base = Yb + slot * 16 * YSTR + m * YSTR + (kc & 7) * 32 + hi * 16;
            v16bf a = ld16(base);
            acc = wmma_bf16(a, Breg[kc], acc);
        }
        __syncthreads();                        // all LDS reads done before overwriting y_{t-2}
        if (hi == 0) {                          // D rows 0..7 = batches
#pragma unroll
            for (int r = 0; r < 8; ++r) {
                size_t offs = ((size_t)r * SEQ + t) * D_OUT + o;
                float y = acc[r] + dreg[r];
                out[offs] = y;
                Yb[(t & 1) * 16 * YSTR + r * YSTR + o] = tobf(y);
            }
            if (t + 1 < SEQ) {                  // prefetch next step's delta
#pragma unroll
                for (int r = 0; r < 8; ++r)
                    dreg[r] = delta[((size_t)r * SEQ + (t + 1)) * D_OUT + o];
            }
        }
        __syncthreads();                        // new y visible to every wave next step
    }
}

// ---------------------------------------------------------------- launcher
extern "C" void kernel_launch(void* const* d_in, const int* in_sizes, int n_in,
                              void* d_out, int out_size, void* d_ws, size_t ws_size,
                              hipStream_t stream) {
    const float* u        = (const float*)d_in[0];
    const float* m_y      = (const float*)d_in[1];
    const float* m_u      = (const float*)d_in[2];
    const float* m_phi    = (const float*)d_in[3];
    const float* eig_vals = (const float*)d_in[4];
    const float* eig_vecs = (const float*)d_in[5];
    float* out = (float*)d_out;
    (void)in_sizes; (void)n_in; (void)out_size; (void)ws_size;

    char* ws = (char*)d_ws;
    size_t off = 0;
    auto alloc = [&](size_t bytes) {
        void* p = ws + off;
        off = (off + bytes + 255) & ~(size_t)255;
        return p;
    };
    bf16*  vS    = (bf16*) alloc((size_t)SEQ * KEIG * 2);
    bf16*  phiT  = (bf16*) alloc((size_t)D_OUT * KDIM * 2);
    bf16*  muB   = (bf16*) alloc((size_t)KU * D_OUT * D_OUT * 2);
    bf16*  myB   = (bf16*) alloc((size_t)KY * D_OUT * D_OUT * 2);
    bf16*  uT    = (bf16*) alloc((size_t)BATCH * D_OUT * SEQ * 2);
    bf16*  uP    = (bf16*) alloc((size_t)BATCH * SEQ * D_OUT * 2);
    float* delta = (float*)alloc((size_t)BATCH * SEQ * D_OUT * 4);
    bf16*  xt    = (bf16*) alloc((size_t)SEQ * KDIM * 2);   // reused per batch

    prep_filters<<<(SEQ * KEIG + 255) / 256, 256, 0, stream>>>(eig_vecs, eig_vals, vS);
    prep_phiT<<<(D_OUT * KDIM + 255) / 256, 256, 0, stream>>>(m_phi, phiT);
    prep_mu<<<(KU * D_OUT * D_OUT + 255) / 256, 256, 0, stream>>>(m_u, muB);
    prep_my<<<(KY * D_OUT * D_OUT + 255) / 256, 256, 0, stream>>>(m_y, myB);
    prep_u<<<dim3(SEQ / 32, D_OUT / 32, BATCH), 256, 0, stream>>>(u, uT, uP);

    for (int b = 0; b < BATCH; ++b) {
        conv_kernel<<<dim3(128, 3), 256, 0, stream>>>(uT + (size_t)b * D_OUT * SEQ, vS, xt);
        proj_kernel<<<64, 256, 0, stream>>>(xt, uP + (size_t)b * SEQ * D_OUT, phiT, muB,
                                            delta + (size_t)b * SEQ * D_OUT);
    }
    recur_kernel<<<1, 512, 0, stream>>>(myB, delta, out);
}
